// GTEATransT2V_VTrain_63144609186277
// MI455X (gfx1250) — matrix operand order repaired and checked by reading, not compiled, not run on hardware
//
#include <hip/hip_runtime.h>

typedef __attribute__((ext_vector_type(16))) _Float16 v16h;
typedef __attribute__((ext_vector_type(8)))  float    v8f;

constexpr int kH = 128, kS = 8, kDh = 32, kEdgeIn = 96, kC = 16;

__device__ __forceinline__ v8f wmma_f16(v16h a, v16h b, v8f c) {
  return __builtin_amdgcn_wmma_f32_16x16x32_f16(false, a, false, b, (short)0, c, false, false);
}

// A operand (16x32 f16) gather from row-major LDS tile, per CDNA5 ISA layout:
// lanes 0-15 hold K in {0..7, 16..23}; lanes 16-31 hold K in {8..15, 24..31}.
__device__ __forceinline__ v16h lds_load_a(const _Float16* sA, int ld, int kt, int lane) {
  const int m  = lane & 15;
  const int kb = (lane & 16) ? 8 : 0;
  const _Float16* p = sA + m * ld + kt * 32 + kb;
  v16h a;
#pragma unroll
  for (int i = 0; i < 16; ++i) a[i] = p[((i >> 3) << 4) + (i & 7)];
  return a;
}

// B operand pre-packed per lane: tile stride = 512 halves, lane stride = 16 halves.
__device__ __forceinline__ v16h load_b_packed(const _Float16* P, int tile, int lane) {
  return ((const v16h*)P)[tile * 32 + lane];
}

// Pack row-major f32 weight [K,N] into WMMA B-operand layout (f16).
// Per ISA: lane n=lane&15 is column, lanes>=16 carry K offset +8, half i -> K = kb + ((i>>3)<<4) + (i&7).
__global__ void pack_weight_kernel(const float* __restrict__ W, _Float16* __restrict__ P,
                                   int K, int N) {
  const int idx = blockIdx.x * blockDim.x + threadIdx.x;
  if (idx >= K * N) return;
  const int i    = idx & 15;
  const int lane = (idx >> 4) & 31;
  const int t    = idx >> 9;
  const int KT   = K >> 5;
  const int kt   = t % KT;
  const int nt   = t / KT;
  const int n    = lane & 15;
  const int kb   = (lane & 16) ? 8 : 0;
  const int k    = kt * 32 + kb + ((i >> 3) << 4) + (i & 7);
  P[idx] = (_Float16)W[k * N + nt * 16 + n];
}

// ---------------- Embed: e = relu(concat(edge_feat, t2v) @ W_ein + b) --------
__global__ __launch_bounds__(128) void embed_kernel(
    const float* __restrict__ edge_features, const float* __restrict__ seq_times,
    const float* __restrict__ t2v_w0, const float* __restrict__ t2v_b0,
    const float* __restrict__ t2v_w, const float* __restrict__ t2v_b,
    const _Float16* __restrict__ Wp, const float* __restrict__ b_ein,
    _Float16* __restrict__ eacts) {
  __shared__ alignas(16) _Float16 sA[4][16 * kH];
  const int w = threadIdx.x >> 5, lane = threadIdx.x & 31;
  const int r0 = (blockIdx.x * 4 + w) * 16;
  const float w0 = t2v_w0[0], b0 = t2v_b0[0];
  for (int mrow = 0; mrow < 16; ++mrow) {
    const int r = r0 + mrow;
    const float t = seq_times[r];
    const float* ef = edge_features + (size_t)r * kEdgeIn;
#pragma unroll
    for (int j = 0; j < 4; ++j) {
      const int col = lane + 32 * j;
      float v;
      if (col < kEdgeIn)       v = ef[col];
      else if (col == kEdgeIn) v = t * w0 + b0;
      else                     v = __sinf(t * t2v_w[col - kEdgeIn - 1] + t2v_b[col - kEdgeIn - 1]);
      sA[w][mrow * kH + col] = (_Float16)v;
    }
  }
  __syncthreads();
  for (int nt = 0; nt < 8; ++nt) {
    v8f c = {};
#pragma unroll
    for (int kt = 0; kt < 4; ++kt)
      c = wmma_f16(lds_load_a(sA[w], kH, kt, lane), load_b_packed(Wp, nt * 4 + kt, lane), c);
    const int n = lane & 15, mo = (lane & 16) ? 8 : 0;
    const float bias = b_ein[nt * 16 + n];
#pragma unroll
    for (int r = 0; r < 8; ++r) {
      float v = c[r] + bias;
      eacts[(size_t)(r0 + mo + r) * kH + nt * 16 + n] = (_Float16)(v > 0.f ? v : 0.f);
    }
  }
}

// ---------------- Transformer encoder layer + masked mean pool ---------------
struct EncP {
  const _Float16 *Wqkv, *Wo, *W1, *W2;                    // packed f16
  const float *bqkv, *bo, *ln1g, *ln1b, *b1, *b2, *ln2g, *ln2b;
};

__device__ __forceinline__ void layernorm16(float* Y, const float* g, const float* b,
                                            int lane, _Float16* Xout) {
  const int row = lane & 15;
  const int h0  = (lane & 16) ? 64 : 0;
  float* y = Y + row * kH;
  float s = 0.f, s2 = 0.f;
#pragma unroll 16
  for (int j = 0; j < 64; ++j) { const float v = y[h0 + j]; s += v; s2 += v * v; }
  s  += __shfl_xor(s, 16, 32);
  s2 += __shfl_xor(s2, 16, 32);
  const float mu  = s * (1.f / kH);
  const float var = s2 * (1.f / kH) - mu * mu;
  const float rs  = rsqrtf(var + 1e-5f);
#pragma unroll 16
  for (int j = 0; j < 64; ++j) {
    const int c = h0 + j;
    const float v = (y[c] - mu) * rs * g[c] + b[c];
    y[c] = v;
    Xout[row * kH + c] = (_Float16)v;
  }
}

__global__ __launch_bounds__(64) void encoder_pool_kernel(
    const _Float16* __restrict__ eacts, const int* __restrict__ edge_len,
    EncP p, _Float16* __restrict__ pooled) {
  __shared__ alignas(16) _Float16 sX[2][16 * kH];        //  8 KB
  __shared__ alignas(16) _Float16 sQKV[2][16 * 3 * kH];  // 24 KB
  __shared__ alignas(16) _Float16 sT[2][16 * kH];        //  8 KB
  __shared__ alignas(16) float    sY[2][16 * kH];        // 16 KB
  const int w = threadIdx.x >> 5;
  const int lane = threadIdx.x & 31;
  const int e0 = blockIdx.x * 4 + w * 2;
  const int r0 = e0 * kS;  // 16 contiguous token rows = 2 edges

  {  // stage x tile (4 KB contiguous)
    const uint4* src = (const uint4*)(eacts + (size_t)r0 * kH);
    uint4* dst = (uint4*)sX[w];
#pragma unroll
    for (int j = 0; j < 8; ++j) dst[lane + 32 * j] = src[lane + 32 * j];
  }
  __syncthreads();

  // QKV = x @ Wqkv + b  (N = 384 -> 24 tiles, 96 WMMAs)
  for (int nt = 0; nt < 24; ++nt) {
    v8f c = {};
#pragma unroll
    for (int kt = 0; kt < 4; ++kt)
      c = wmma_f16(lds_load_a(sX[w], kH, kt, lane), load_b_packed(p.Wqkv, nt * 4 + kt, lane), c);
    const int n = lane & 15, mo = (lane & 16) ? 8 : 0;
    const float bias = p.bqkv[nt * 16 + n];
#pragma unroll
    for (int r = 0; r < 8; ++r)
      sQKV[w][(mo + r) * (3 * kH) + nt * 16 + n] = (_Float16)(c[r] + bias);
  }
  __syncthreads();

  // masked MHA (S=8, dh=32) on VALU: lane = (head, q-row)
  {
    const int hh = lane >> 3, q = lane & 7;
    const float inv = 0.17677669529663687f;  // 1/sqrt(32)
    for (int le = 0; le < 2; ++le) {
      const int len = edge_len[e0 + le];
      const _Float16* Qp = sQKV[w] + (le * 8 + q) * (3 * kH) + hh * kDh;
      float sc[8];
#pragma unroll
      for (int k = 0; k < 8; ++k) {
        const _Float16* Kp = sQKV[w] + (le * 8 + k) * (3 * kH) + kH + hh * kDh;
        float s = 0.f;
#pragma unroll
        for (int d = 0; d < kDh; ++d) s += (float)Qp[d] * (float)Kp[d];
        sc[k] = (k < len) ? s * inv : -1e9f;
      }
      float mx = sc[0];
#pragma unroll
      for (int k = 1; k < 8; ++k) mx = fmaxf(mx, sc[k]);
      float den = 0.f;
#pragma unroll
      for (int k = 0; k < 8; ++k) { sc[k] = __expf(sc[k] - mx); den += sc[k]; }
      const float rd = 1.f / den;
#pragma unroll
      for (int d = 0; d < kDh; ++d) {
        float o = 0.f;
#pragma unroll
        for (int k = 0; k < 8; ++k)
          o += sc[k] * (float)sQKV[w][(le * 8 + k) * (3 * kH) + 2 * kH + hh * kDh + d];
        sT[w][(le * 8 + q) * kH + hh * kDh + d] = (_Float16)(o * rd);
      }
    }
  }
  __syncthreads();

  // y = x + o @ Wo + bo ; LN1
  for (int nt = 0; nt < 8; ++nt) {
    v8f c = {};
#pragma unroll
    for (int kt = 0; kt < 4; ++kt)
      c = wmma_f16(lds_load_a(sT[w], kH, kt, lane), load_b_packed(p.Wo, nt * 4 + kt, lane), c);
    const int n = lane & 15, mo = (lane & 16) ? 8 : 0;
    const float bias = p.bo[nt * 16 + n];
#pragma unroll
    for (int r = 0; r < 8; ++r) {
      const int row = mo + r, col = nt * 16 + n;
      sY[w][row * kH + col] = c[r] + bias + (float)sX[w][row * kH + col];
    }
  }
  __syncthreads();
  layernorm16(sY[w], p.ln1g, p.ln1b, lane, sX[w]);
  __syncthreads();

  // h1 = relu(x @ W1 + b1)
  for (int nt = 0; nt < 8; ++nt) {
    v8f c = {};
#pragma unroll
    for (int kt = 0; kt < 4; ++kt)
      c = wmma_f16(lds_load_a(sX[w], kH, kt, lane), load_b_packed(p.W1, nt * 4 + kt, lane), c);
    const int n = lane & 15, mo = (lane & 16) ? 8 : 0;
    const float bias = p.b1[nt * 16 + n];
#pragma unroll
    for (int r = 0; r < 8; ++r) {
      const float v = c[r] + bias;
      sT[w][(mo + r) * kH + nt * 16 + n] = (_Float16)(v > 0.f ? v : 0.f);
    }
  }
  __syncthreads();

  // y = y + h1 @ W2 + b2 ; LN2
  for (int nt = 0; nt < 8; ++nt) {
    v8f c = {};
#pragma unroll
    for (int kt = 0; kt < 4; ++kt)
      c = wmma_f16(lds_load_a(sT[w], kH, kt, lane), load_b_packed(p.W2, nt * 4 + kt, lane), c);
    const int n = lane & 15, mo = (lane & 16) ? 8 : 0;
    const float bias = p.b2[nt * 16 + n];
#pragma unroll
    for (int r = 0; r < 8; ++r)
      sY[w][(mo + r) * kH + nt * 16 + n] += c[r] + bias;
  }
  __syncthreads();
  layernorm16(sY[w], p.ln2g, p.ln2b, lane, sX[w]);
  __syncthreads();

  // masked mean pool over S
  for (int le = 0; le < 2; ++le) {
    const int len = edge_len[e0 + le];
    const float invl = 1.f / (float)len;
#pragma unroll
    for (int j = 0; j < 4; ++j) {
      const int col = lane * 4 + j;
      float s = 0.f;
#pragma unroll
      for (int t = 0; t < kS; ++t)
        if (t < len) s += sY[w][(le * 8 + t) * kH + col];
      pooled[(size_t)(e0 + le) * kH + col] = (_Float16)(s * invl);
    }
  }
}

// ------------- m = relu(concat(node_feat[src], e_out) @ W_eout + b) ----------
__global__ __launch_bounds__(128) void m_kernel(
    const float* __restrict__ node_features, const int* __restrict__ edge_src,
    const _Float16* __restrict__ eout, const _Float16* __restrict__ Wp,
    const float* __restrict__ b_eout, float* __restrict__ m_out) {
  __shared__ alignas(16) _Float16 sA[4][16 * 256];
  const int w = threadIdx.x >> 5, lane = threadIdx.x & 31;
  const int r0 = (blockIdx.x * 4 + w) * 16;
  for (int mrow = 0; mrow < 16; ++mrow) {
    const int r = r0 + mrow;
    const int src = edge_src[r];
#pragma unroll
    for (int j = 0; j < 4; ++j) {
      const int c = lane + 32 * j;
      sA[w][mrow * 256 + c]       = (_Float16)node_features[(size_t)src * kH + c];
      sA[w][mrow * 256 + 128 + c] = eout[(size_t)r * kH + c];
    }
  }
  __syncthreads();
  for (int nt = 0; nt < 8; ++nt) {
    v8f c = {};
#pragma unroll
    for (int kt = 0; kt < 8; ++kt)
      c = wmma_f16(lds_load_a(sA[w], 256, kt, lane), load_b_packed(Wp, nt * 8 + kt, lane), c);
    const int n = lane & 15, mo = (lane & 16) ? 8 : 0;
    const float bias = b_eout[nt * 16 + n];
#pragma unroll
    for (int r = 0; r < 8; ++r) {
      const float v = c[r] + bias;
      m_out[(size_t)(r0 + mo + r) * kH + nt * 16 + n] = v > 0.f ? v : 0.f;
    }
  }
}

// ------------- a = leaky_relu(a_vec @ W_attn) --------------------------------
__global__ void a_kernel(const _Float16* __restrict__ avec, const float* __restrict__ W_attn,
                         float* __restrict__ a_out, int E) {
  const int i = blockIdx.x * blockDim.x + threadIdx.x;
  if (i >= E) return;
  const _Float16* row = avec + (size_t)i * kH;
  float s = 0.f;
#pragma unroll 16
  for (int c = 0; c < kH; ++c) s += (float)row[c] * W_attn[c];
  a_out[i] = (s > 0.f) ? s : 0.01f * s;
}

// ------------- segment bounds (edge_dst is sorted) ---------------------------
__global__ void seg_kernel(const int* __restrict__ edge_dst, int* __restrict__ seg_start,
                           int* __restrict__ seg_end, int E) {
  const int i = blockIdx.x * blockDim.x + threadIdx.x;
  if (i >= E) return;
  const int d = edge_dst[i];
  if (i == 0 || edge_dst[i - 1] != d) seg_start[d] = i;
  if (i == E - 1 || edge_dst[i + 1] != d) seg_end[d] = i + 1;
}

// ------------- deterministic segment softmax + weighted sum ------------------
__global__ __launch_bounds__(256) void agg_kernel(
    const float* __restrict__ a, const float* __restrict__ m,
    const int* __restrict__ seg_start, const int* __restrict__ seg_end,
    float* __restrict__ h_agg, int ndst) {
  const int w = blockIdx.x * (blockDim.x >> 5) + (threadIdx.x >> 5);
  const int lane = threadIdx.x & 31;
  if (w >= ndst) return;
  const int s0 = seg_start[w], s1 = seg_end[w];
  float mx = -1e30f;
  for (int i = s0 + lane; i < s1; i += 32) mx = fmaxf(mx, a[i]);
#pragma unroll
  for (int o = 16; o > 0; o >>= 1) mx = fmaxf(mx, __shfl_xor(mx, o, 32));
  float den = 0.f;
  for (int i = s0 + lane; i < s1; i += 32) den += __expf(a[i] - mx);
#pragma unroll
  for (int o = 16; o > 0; o >>= 1) den += __shfl_xor(den, o, 32);
  const float rd = 1.f / den;
  float acc[4] = {0.f, 0.f, 0.f, 0.f};
  for (int i = s0; i < s1; ++i) {
    const float wgt = __expf(a[i] - mx) * rd;
    const float* mr = m + (size_t)i * kH + lane * 4;
#pragma unroll
    for (int j = 0; j < 4; ++j) acc[j] += wgt * mr[j];
  }
  float* out = h_agg + (size_t)w * kH + lane * 4;
#pragma unroll
  for (int j = 0; j < 4; ++j) out[j] = acc[j];
}

// ------------- node update + final fc ----------------------------------------
__global__ __launch_bounds__(32) void node_kernel(
    const float* __restrict__ node_features, const int* __restrict__ dst_in_src,
    const float* __restrict__ h_agg,
    const _Float16* __restrict__ Weout_p, const float* __restrict__ b_eout,
    const _Float16* __restrict__ Wnode_p, const float* __restrict__ b_node,
    const _Float16* __restrict__ Wfc_p, const float* __restrict__ b_fc,
    float* __restrict__ out, int ndst) {
  __shared__ alignas(16) _Float16 sS[16 * kH];
  __shared__ alignas(16) _Float16 sA2[16 * 256];
  __shared__ alignas(16) _Float16 sH[16 * kH];
  const int lane = threadIdx.x;
  const int d0 = blockIdx.x * 16;
  for (int mrow = 0; mrow < 16; ++mrow) {
    const int d = d0 + mrow;
    const int src = (d < ndst) ? dst_in_src[d] : dst_in_src[0];
#pragma unroll
    for (int j = 0; j < 4; ++j) {
      const int c = lane + 32 * j;
      const _Float16 v = (_Float16)node_features[(size_t)src * kH + c];
      sS[mrow * kH + c] = v;
      sA2[mrow * 256 + c] = v;
    }
  }
  __syncthreads();
  // self_h_tmp = self_h @ W_eout[:128] + b_eout ; sA2[:,128:] = h_agg - tmp
  for (int nt = 0; nt < 8; ++nt) {
    v8f c = {};
#pragma unroll
    for (int kt = 0; kt < 4; ++kt)  // top 128 K rows of packed W_eout (KT=8)
      c = wmma_f16(lds_load_a(sS, kH, kt, lane), load_b_packed(Weout_p, nt * 8 + kt, lane), c);
    const int n = lane & 15, mo = (lane & 16) ? 8 : 0;
    const float bias = b_eout[nt * 16 + n];
#pragma unroll
    for (int r = 0; r < 8; ++r) {
      const int row = mo + r, dd = d0 + row;
      const float hv = (dd < ndst) ? h_agg[(size_t)dd * kH + nt * 16 + n] : 0.f;
      sA2[row * 256 + 128 + nt * 16 + n] = (_Float16)(hv - (c[r] + bias));
    }
  }
  __syncthreads();
  // h = relu(concat @ W_node + b_node)
  for (int nt = 0; nt < 8; ++nt) {
    v8f c = {};
#pragma unroll
    for (int kt = 0; kt < 8; ++kt)
      c = wmma_f16(lds_load_a(sA2, 256, kt, lane), load_b_packed(Wnode_p, nt * 8 + kt, lane), c);
    const int n = lane & 15, mo = (lane & 16) ? 8 : 0;
    const float bias = b_node[nt * 16 + n];
#pragma unroll
    for (int r = 0; r < 8; ++r) {
      const float v = c[r] + bias;
      sH[(mo + r) * kH + nt * 16 + n] = (_Float16)(v > 0.f ? v : 0.f);
    }
  }
  __syncthreads();
  // out = h @ W_fc + b_fc  (single N tile of 16)
  {
    v8f c = {};
#pragma unroll
    for (int kt = 0; kt < 4; ++kt)
      c = wmma_f16(lds_load_a(sH, kH, kt, lane), load_b_packed(Wfc_p, kt, lane), c);
    const int n = lane & 15, mo = (lane & 16) ? 8 : 0;
    const float bias = b_fc[n];
#pragma unroll
    for (int r = 0; r < 8; ++r) {
      const int dd = d0 + mo + r;
      if (dd < ndst) out[(size_t)dd * kC + n] = c[r] + bias;
    }
  }
}

// ============================================================================
extern "C" void kernel_launch(void* const* d_in, const int* in_sizes, int n_in,
                              void* d_out, int out_size, void* d_ws, size_t ws_size,
                              hipStream_t stream) {
  (void)n_in; (void)out_size; (void)ws_size;
  const float* node_features = (const float*)d_in[0];
  const float* edge_features = (const float*)d_in[1];
  const float* seq_times     = (const float*)d_in[2];
  const int*   edge_len      = (const int*)d_in[3];
  const int*   edge_src      = (const int*)d_in[4];
  const int*   edge_dst      = (const int*)d_in[5];
  const int*   dst_in_src    = (const int*)d_in[6];

  const int E    = in_sizes[3];
  const int NDST = in_sizes[6];
  const int R    = E * kS;

  char* wp = (char*)d_ws;
  auto alloc = [&](size_t bytes) -> char* {
    char* r = wp; wp += (bytes + 255) & ~(size_t)255; return r;
  };

  // packed f16 weights
  _Float16* pW_ein  = (_Float16*)alloc((size_t)128 * 128 * 2);
  _Float16* pWqkv_e = (_Float16*)alloc((size_t)128 * 384 * 2);
  _Float16* pWo_e   = (_Float16*)alloc((size_t)128 * 128 * 2);
  _Float16* pW1_e   = (_Float16*)alloc((size_t)128 * 128 * 2);
  _Float16* pW2_e   = (_Float16*)alloc((size_t)128 * 128 * 2);
  _Float16* pWqkv_a = (_Float16*)alloc((size_t)128 * 384 * 2);
  _Float16* pWo_a   = (_Float16*)alloc((size_t)128 * 128 * 2);
  _Float16* pW1_a   = (_Float16*)alloc((size_t)128 * 128 * 2);
  _Float16* pW2_a   = (_Float16*)alloc((size_t)128 * 128 * 2);
  _Float16* pW_eout = (_Float16*)alloc((size_t)256 * 128 * 2);
  _Float16* pW_node = (_Float16*)alloc((size_t)256 * 128 * 2);
  _Float16* pW_fc   = (_Float16*)alloc((size_t)128 * 16 * 2);
  // activations / scratch
  _Float16* e_buf    = (_Float16*)alloc((size_t)R * kH * 2);
  _Float16* eout_buf = (_Float16*)alloc((size_t)E * kH * 2);
  _Float16* avec_buf = (_Float16*)alloc((size_t)E * kH * 2);
  float*    m_buf    = (float*)alloc((size_t)E * kH * 4);
  float*    a_buf    = (float*)alloc((size_t)E * 4);
  int*      seg0     = (int*)alloc((size_t)NDST * 4);
  int*      seg1     = (int*)alloc((size_t)NDST * 4);
  float*    hagg_buf = (float*)alloc((size_t)NDST * kH * 4);

  auto pack = [&](const void* W, _Float16* P, int K, int N) {
    const int total = K * N;
    pack_weight_kernel<<<(total + 255) / 256, 256, 0, stream>>>((const float*)W, P, K, N);
  };
  pack(d_in[11], pW_ein, 128, 128);
  pack(d_in[13], pWqkv_e, 128, 384);
  pack(d_in[15], pWo_e, 128, 128);
  pack(d_in[19], pW1_e, 128, 128);
  pack(d_in[21], pW2_e, 128, 128);
  pack(d_in[25], pWqkv_a, 128, 384);
  pack(d_in[27], pWo_a, 128, 128);
  pack(d_in[31], pW1_a, 128, 128);
  pack(d_in[33], pW2_a, 128, 128);
  pack(d_in[38], pW_eout, 256, 128);
  pack(d_in[40], pW_node, 256, 128);
  pack(d_in[42], pW_fc, 128, 16);

  // e = relu(concat @ W_ein + b_ein)
  embed_kernel<<<R / 64, 128, 0, stream>>>(
      edge_features, seq_times,
      (const float*)d_in[7], (const float*)d_in[8], (const float*)d_in[9], (const float*)d_in[10],
      pW_ein, (const float*)d_in[12], e_buf);

  // two encoder+pool passes
  EncP pe{pWqkv_e, pWo_e, pW1_e, pW2_e,
          (const float*)d_in[14], (const float*)d_in[16], (const float*)d_in[17],
          (const float*)d_in[18], (const float*)d_in[20], (const float*)d_in[22],
          (const float*)d_in[23], (const float*)d_in[24]};
  EncP pa{pWqkv_a, pWo_a, pW1_a, pW2_a,
          (const float*)d_in[26], (const float*)d_in[28], (const float*)d_in[29],
          (const float*)d_in[30], (const float*)d_in[32], (const float*)d_in[34],
          (const float*)d_in[35], (const float*)d_in[36]};
  encoder_pool_kernel<<<E / 4, 64, 0, stream>>>(e_buf, edge_len, pe, eout_buf);
  encoder_pool_kernel<<<E / 4, 64, 0, stream>>>(e_buf, edge_len, pa, avec_buf);

  m_kernel<<<E / 64, 128, 0, stream>>>(node_features, edge_src, eout_buf, pW_eout,
                                       (const float*)d_in[39], m_buf);
  a_kernel<<<(E + 255) / 256, 256, 0, stream>>>(avec_buf, (const float*)d_in[37], a_buf, E);
  seg_kernel<<<(E + 255) / 256, 256, 0, stream>>>(edge_dst, seg0, seg1, E);
  agg_kernel<<<(NDST + 7) / 8, 256, 0, stream>>>(a_buf, m_buf, seg0, seg1, hagg_buf, NDST);
  node_kernel<<<(NDST + 15) / 16, 32, 0, stream>>>(
      node_features, dst_in_src, hagg_buf,
      pW_eout, (const float*)d_in[39], pW_node, (const float*)d_in[41],
      pW_fc, (const float*)d_in[43], (float*)d_out, NDST);
}
